// Head_35880156791224
// MI455X (gfx1250) — matrix operand order repaired and checked by reading, not compiled, not run on hardware
//
#include <hip/hip_runtime.h>
#include <hip/hip_bf16.h>
#include <math.h>

// Problem constants (match reference)
#define B_  8
#define T_  4096
#define C_  32
#define HS_ 64

// Attention tiling
#define BM   128   // rows per workgroup (8 waves x 16 rows)
#define BN   64    // key/value columns per tile
#define NW   8     // waves per workgroup
#define PADE 8     // bf16 pad: (64+8)*2 = 144 B row stride, 16B aligned
#define LDK  (HS_ + PADE)
#define LDV  (BN + PADE)

#define SCALE 0.1767766952966369f  // C^-0.5 = 1/sqrt(32)

typedef __attribute__((ext_vector_type(16))) __bf16 v16bf;
typedef __attribute__((ext_vector_type(8)))  float  v8f;
typedef __attribute__((ext_vector_type(8)))  unsigned int v8u;
typedef __attribute__((ext_vector_type(4)))  float  f4;

// A-matrix (16x32 bf16) K index for VGPR pair v, lane-half h (contiguous: 16h + 2v).
static __device__ __forceinline__ int ka_idx(int v, int half) {
  return 2 * v + 16 * half;
}

static __device__ __forceinline__ v8f zero8() {
  v8f z = {0.f, 0.f, 0.f, 0.f, 0.f, 0.f, 0.f, 0.f};
  return z;
}

static __device__ __forceinline__ v8f wmma_bf16(v16bf a, v16bf b, v8f c) {
  return __builtin_amdgcn_wmma_f32_16x16x32_bf16(
      false, a, false, b, (short)0, c, false, false);
}

// ---- CDNA5 async global->LDS copy (ASYNCcnt path), inline asm ------------
// LDS destination VGPR holds the workgroup-relative LDS byte offset, which is
// the low 32 bits of a generic pointer to __shared__ (aperture maps addr[31:0]).
static __device__ __forceinline__ void async_ld_b128(void* lds_ptr, const void* gptr) {
  unsigned lds_off = (unsigned)(unsigned long long)lds_ptr;
  unsigned long long g = (unsigned long long)gptr;
  asm volatile("global_load_async_to_lds_b128 %0, %1, off"
               :: "v"(lds_off), "v"(g) : "memory");
}
static __device__ __forceinline__ void wait_asynccnt0() {
  asm volatile("s_wait_asynccnt 0x0" ::: "memory");
}

// Copy one 64x64 bf16 K tile (8 KB) with 512 x 16B async chunks (2/thread).
static __device__ __forceinline__ void async_copy_k_tile(
    const __bf16* kb, int s0, __bf16 (*dst)[LDK], int tid) {
#pragma unroll
  for (int c = 0; c < 2; ++c) {
    const int chunk = tid + 256 * c;      // 0..511
    const int row   = chunk >> 3;         // 8 x 16B chunks per 128B row
    const int off   = (chunk & 7) * 8;    // element offset
    async_ld_b128(&dst[row][off], kb + (size_t)(s0 + row) * HS_ + off);
  }
}

// V tile: register prefetch (32B/thread) + LDS transpose store.
static __device__ __forceinline__ v8u load_v_chunk(const __bf16* vb, int s0, int tid) {
  const int row = tid >> 2;
  const int hs  = (tid & 3) << 4;         // 16-element (32B) segment
  return *(const v8u*)(vb + (size_t)(s0 + row) * HS_ + hs);
}
static __device__ __forceinline__ void tstore_v(v8u vv, __bf16 (*vt)[LDV], int tid) {
  const int row = tid >> 2;
  const int hs  = (tid & 3) << 4;
  __bf16 tmp[16];
  *(v8u*)tmp = vv;
#pragma unroll
  for (int e = 0; e < 16; ++e) vt[hs + e][row] = tmp[e];
}

// ---------------------------------------------------------------------------
// Kernel 1: q/k/v = x @ W{q,k,v}  (f32 in -> bf16 out), WMMA 16x16x32 bf16.
// ---------------------------------------------------------------------------
__global__ __launch_bounds__(32) void proj_qkv(
    const float* __restrict__ x, const float* __restrict__ Wq,
    const float* __restrict__ Wk, const float* __restrict__ Wv,
    __bf16* __restrict__ qb, __bf16* __restrict__ kb, __bf16* __restrict__ vb)
{
  __shared__ __bf16 Wt[3][HS_][C_ + PADE];  // transposed [h][c]

  const int lane = threadIdx.x & 31;
  const int half = lane >> 4;
  const int ln   = lane & 15;
  const int row0 = blockIdx.x * 16;

  const float* Ws[3] = {Wq, Wk, Wv};
#pragma unroll
  for (int w = 0; w < 3; ++w) {
    for (int i = lane; i < C_ * HS_; i += 32) {
      const int c = i >> 6;
      const int h = i & 63;
      Wt[w][h][c] = (__bf16)Ws[w][i];
    }
  }
  __syncthreads();

  v16bf xa;
#pragma unroll
  for (int v = 0; v < 8; ++v) {
    const int K = ka_idx(v, half);
    const float* xr = x + (size_t)(row0 + ln) * C_;
    xa[2 * v]     = (__bf16)xr[K];
    xa[2 * v + 1] = (__bf16)xr[K + 1];
  }

  __bf16* outs[3] = {qb, kb, vb};
#pragma unroll
  for (int w = 0; w < 3; ++w) {
#pragma unroll
    for (int n = 0; n < 4; ++n) {
      v16bf bw;
#pragma unroll
      for (int v = 0; v < 8; ++v) {
        const int K = 16 * half + 2 * v;
        bw[2 * v]     = Wt[w][16 * n + ln][K];
        bw[2 * v + 1] = Wt[w][16 * n + ln][K + 1];
      }
      v8f acc = zero8();
      acc = wmma_bf16(xa, bw, acc);
#pragma unroll
      for (int r = 0; r < 8; ++r) {
        const int grow = row0 + r + 8 * half;
        outs[w][(size_t)grow * HS_ + 16 * n + ln] = (__bf16)acc[r];
      }
    }
  }
}

// ---------------------------------------------------------------------------
// Kernel 2: causal attention, double-buffered async K tiles + prefetched V.
// ---------------------------------------------------------------------------
__global__ __launch_bounds__(256, 2) void attn_fwd(
    const __bf16* __restrict__ qg, const __bf16* __restrict__ kg,
    const __bf16* __restrict__ vg, float* __restrict__ og,
    float* __restrict__ attg)
{
  __shared__ __bf16 Kt[2][BN][LDK];        // K tiles, [s][h], double buffered
  __shared__ __bf16 Vt[2][HS_][LDV];       // V tiles transposed, [h][s]
  __shared__ __bf16 Pb[NW][16][LDV];       // per-wave P staging (C->A relayout)

  const int tid  = threadIdx.x;
  const int wave = tid >> 5;
  const int lane = tid & 31;
  const int half = lane >> 4;
  const int ln   = lane & 15;

  const int nblk = T_ / BM;
  const int b    = blockIdx.x / nblk;
  const int m0   = (blockIdx.x % nblk) * BM;
  const int mrow = m0 + wave * 16;

  const __bf16* qb = qg + (size_t)b * T_ * HS_;
  const __bf16* kb = kg + (size_t)b * T_ * HS_;
  const __bf16* vb = vg + (size_t)b * T_ * HS_;
  float* attb = attg + (size_t)b * T_ * T_;

  // Q (16 x 64) as two 16x32 bf16 A-fragments, register resident
  v16bf qa[2];
#pragma unroll
  for (int f = 0; f < 2; ++f) {
#pragma unroll
    for (int v = 0; v < 8; ++v) {
      const int K = 32 * f + ka_idx(v, half);
      const __bf16* qr = qb + (size_t)(mrow + ln) * HS_;
      qa[f][2 * v]     = qr[K];
      qa[f][2 * v + 1] = qr[K + 1];
    }
  }

  const int jmax = (m0 + BM - 1) / BN;  // last causal K tile (WG-uniform), >= 1

  float mrun[8], lrun[8];
#pragma unroll
  for (int r = 0; r < 8; ++r) { mrun[r] = -INFINITY; lrun[r] = 0.f; }

  // ---------------- Pass 1: softmax statistics ----------------
  async_copy_k_tile(kb, 0, Kt[0], tid);
  for (int j = 0; j <= jmax; ++j) {
    const int cur = j & 1;
    wait_asynccnt0();          // our slice of tile j landed
    __syncthreads();           // everyone's slice landed; buf cur^1 now free
    if (j < jmax) async_copy_k_tile(kb, (j + 1) * BN, Kt[cur ^ 1], tid);

    const int s0 = j * BN;
#pragma unroll
    for (int n = 0; n < 4; ++n) {
      v8f acc = zero8();
#pragma unroll
      for (int f = 0; f < 2; ++f) {
        v16bf bk;
#pragma unroll
        for (int v = 0; v < 8; ++v) {
          const int K = 32 * f + 16 * half + 2 * v;  // B frag: K=h, N=s
          bk[2 * v]     = Kt[cur][16 * n + ln][K];
          bk[2 * v + 1] = Kt[cur][16 * n + ln][K + 1];
        }
        acc = wmma_bf16(qa[f], bk, acc);
      }
      const int scol = s0 + 16 * n + ln;
#pragma unroll
      for (int r = 0; r < 8; ++r) {
        const int grow = mrow + r + 8 * half;
        float sv = (scol <= grow) ? acc[r] * SCALE : -INFINITY;
        float cm = sv;
#pragma unroll
        for (int off = 1; off < 16; off <<= 1)
          cm = fmaxf(cm, __shfl_xor(cm, off, 32));
        const float mn = fmaxf(mrun[r], cm);
        float cs = __expf(sv - mn);  // exp(-inf) == 0 handles the mask
#pragma unroll
        for (int off = 1; off < 16; off <<= 1)
          cs += __shfl_xor(cs, off, 32);
        lrun[r] = lrun[r] * __expf(mrun[r] - mn) + cs;
        mrun[r] = mn;
      }
    }
  }

  float linv[8];
#pragma unroll
  for (int r = 0; r < 8; ++r) linv[r] = 1.f / lrun[r];

  __syncthreads();  // pass-1 readers done before pass-2 prologue reuses Kt[0]

  // ---------------- Pass 2: write P, accumulate O = P @ V ----------------
  v8f oacc[4];
#pragma unroll
  for (int h = 0; h < 4; ++h) oacc[h] = zero8();

  async_copy_k_tile(kb, 0, Kt[0], tid);
  {
    v8u v0 = load_v_chunk(vb, 0, tid);
    tstore_v(v0, Vt[0], tid);
  }
  v8u vreg = load_v_chunk(vb, BN, tid);  // tile 1 (jmax >= 1 always)

  for (int j = 0; j <= jmax; ++j) {
    const int cur = j & 1;
    const int nxt = cur ^ 1;
    wait_asynccnt0();
    __syncthreads();  // Kt[cur]/Vt[cur] visible; nxt buffers free
    if (j < jmax) {
      async_copy_k_tile(kb, (j + 1) * BN, Kt[nxt], tid);
      tstore_v(vreg, Vt[nxt], tid);                         // tile j+1
      if (j + 1 < jmax) vreg = load_v_chunk(vb, (j + 2) * BN, tid);
    }

    const int s0 = j * BN;
#pragma unroll
    for (int n = 0; n < 4; ++n) {
      v8f acc = zero8();
#pragma unroll
      for (int f = 0; f < 2; ++f) {
        v16bf bk;
#pragma unroll
        for (int v = 0; v < 8; ++v) {
          const int K = 32 * f + 16 * half + 2 * v;
          bk[2 * v]     = Kt[cur][16 * n + ln][K];
          bk[2 * v + 1] = Kt[cur][16 * n + ln][K + 1];
        }
        acc = wmma_bf16(qa[f], bk, acc);
      }
      const int scol = s0 + 16 * n + ln;
#pragma unroll
      for (int r = 0; r < 8; ++r) {
        const int grow = mrow + r + 8 * half;
        const float sv = (scol <= grow) ? acc[r] * SCALE : -INFINITY;
        const float p  = __expf(sv - mrun[r]) * linv[r];    // masked -> 0
        __builtin_nontemporal_store(p, &attb[(size_t)grow * T_ + scol]);
        Pb[wave][r + 8 * half][16 * n + ln] = (__bf16)p;    // C-layout -> LDS
      }
    }

    // O += P (16x64) @ V (64x64); Pb/Vt[cur] reads are same-wave in-order LDS
#pragma unroll
    for (int f = 0; f < 2; ++f) {
      v16bf pa;
#pragma unroll
      for (int v = 0; v < 8; ++v) {
        const int K = 32 * f + ka_idx(v, half);
        pa[2 * v]     = Pb[wave][ln][K];
        pa[2 * v + 1] = Pb[wave][ln][K + 1];
      }
#pragma unroll
      for (int h = 0; h < 4; ++h) {
        v16bf vf;
#pragma unroll
        for (int v = 0; v < 8; ++v) {
          const int K = 32 * f + 16 * half + 2 * v;  // B frag: K=s, N=h
          vf[2 * v]     = Vt[cur][16 * h + ln][K];
          vf[2 * v + 1] = Vt[cur][16 * h + ln][K + 1];
        }
        oacc[h] = wmma_bf16(pa, vf, oacc[h]);
      }
    }
  }

  // store O (f32)
#pragma unroll
  for (int h = 0; h < 4; ++h) {
#pragma unroll
    for (int r = 0; r < 8; ++r) {
      const int grow = mrow + r + 8 * half;
      og[((size_t)b * T_ + grow) * HS_ + 16 * h + ln] = oacc[h][r];
    }
  }

  // zero-fill strictly-future columns [m0+BM, T) for this block's rows (NT)
  const int c0 = m0 + BM;
  if (c0 < T_) {
    const int n4 = (T_ - c0) >> 2;
    const f4 z4 = {0.f, 0.f, 0.f, 0.f};
    for (int row = 0; row < BM; ++row) {
      f4* rp = (f4*)(attb + (size_t)(m0 + row) * T_ + c0);
      for (int i = tid; i < n4; i += 256)
        __builtin_nontemporal_store(z4, rp + i);
    }
  }
}

// ---------------------------------------------------------------------------
// Launch: inputs [x, Wq, Wk, Wv] (f32); d_out = [output (B*T*HS) | attention
// (B*T*T)] f32; d_ws holds bf16 q/k/v (12 MB).
// ---------------------------------------------------------------------------
extern "C" void kernel_launch(void* const* d_in, const int* in_sizes, int n_in,
                              void* d_out, int out_size, void* d_ws,
                              size_t ws_size, hipStream_t stream) {
  (void)in_sizes; (void)n_in; (void)out_size; (void)ws_size;
  const float* x  = (const float*)d_in[0];
  const float* Wq = (const float*)d_in[1];
  const float* Wk = (const float*)d_in[2];
  const float* Wv = (const float*)d_in[3];

  float* out_o   = (float*)d_out;
  float* out_att = out_o + (size_t)B_ * T_ * HS_;

  const size_t nqkv = (size_t)B_ * T_ * HS_;
  __bf16* qb = (__bf16*)d_ws;
  __bf16* kb = qb + nqkv;
  __bf16* vb = kb + nqkv;

  proj_qkv<<<(B_ * T_) / 16, 32, 0, stream>>>(x, Wq, Wk, Wv, qb, kb, vb);
  attn_fwd<<<B_ * (T_ / BM), 256, 0, stream>>>(qb, kb, vb, out_o, out_att);
}
